// CausalAttn_26182120636802
// MI455X (gfx1250) — compile-verified, matrix operand order checked by reading, not run on hardware
//
#include <hip/hip_runtime.h>

typedef __bf16 bf16;
typedef __attribute__((ext_vector_type(16))) __bf16 v16bf;
typedef __attribute__((ext_vector_type(8)))  __bf16 v8bf;
typedef __attribute__((ext_vector_type(4)))  __bf16 v4bf;
typedef __attribute__((ext_vector_type(8)))  float  v8f;

#define L_SEQ   2048
#define DMODEL  2048
#define NHEAD   16
#define DHEAD   128
#define NBATCH  2
#define LOG2_THETA 18.931568569324174f   /* log2(500000) */
#define QSCALE     0.08838834764831845f  /* 1/sqrt(128)  */
#define LOG2E      1.4426950408889634f

__device__ __forceinline__ float fexp2(float x) {
#if __has_builtin(__builtin_amdgcn_exp2f)
  return __builtin_amdgcn_exp2f(x);
#else
  return exp2f(x);
#endif
}
__device__ __forceinline__ float frcp(float x) {
#if __has_builtin(__builtin_amdgcn_rcpf)
  return __builtin_amdgcn_rcpf(x);
#else
  return 1.0f / x;
#endif
}

// Async 16-byte global -> LDS copy (CDNA5 GLOBAL_LOAD_ASYNC_TO_LDS_B128).
// Per lane: LDS[vdst] = MEM[vaddr], 16B; tracked by ASYNCcnt.
// LDS operand = low 32 bits of the flat pointer (LDS aperture maps addr[31:0]).
__device__ __forceinline__ void async_copy16(const bf16* g, bf16* l) {
  unsigned lo = (unsigned)(unsigned long long)l;
  unsigned long long ga = (unsigned long long)g;
  asm volatile("global_load_async_to_lds_b128 %0, %1, off"
               :: "v"(lo), "v"(ga) : "memory");
}
__device__ __forceinline__ void wait_async() {
  asm volatile("s_wait_asynccnt 0x0" ::: "memory");
}

// K-index inside a 16x32 bf16 A-fragment (or col-major B-fragment) for
// vector element e (0..15) and lane-half, per the CDNA5 16-bit A layout.
// e=0..7 -> k = 8*half + 0..7 ; e=8..15 -> k = 16 + 8*half + 0..7
// (two contiguous 16B runs per lane -> ds_load_b128 pairs)
__device__ __forceinline__ int frag_k(int e, int half) {
  int v = e >> 1;
  int base = (v < 4) ? (v << 1) : (16 + ((v - 4) << 1));
  return base + (e & 1) + (half << 3);
}

__device__ __forceinline__ v8f wmma_bf16(v16bf a, v16bf b, v8f c) {
  return __builtin_amdgcn_wmma_f32_16x16x32_bf16(false, a, false, b,
                                                 (short)0, c, false, false);
}

// ---------------------------------------------------------------------------
// fp32 -> bf16 flat conversion (x)
// ---------------------------------------------------------------------------
__global__ __launch_bounds__(256) void cvt_f32_bf16(const float* __restrict__ s,
                                                    bf16* __restrict__ d, int n) {
  int i = (blockIdx.x * blockDim.x + threadIdx.x) * 4;
  if (i + 3 < n) {
    float4 v = *(const float4*)(s + i);
    d[i + 0] = (bf16)v.x; d[i + 1] = (bf16)v.y;
    d[i + 2] = (bf16)v.z; d[i + 3] = (bf16)v.w;
  }
}

// ---------------------------------------------------------------------------
// fp32 W(K,N) -> bf16 W^T(N,K), 32x32 tiles through LDS.  grid (64,64).
// Puts weights K-contiguous so WMMA B fragments read as b128 from LDS.
// ---------------------------------------------------------------------------
__global__ __launch_bounds__(256) void cvt_w_transpose(
    const float* __restrict__ src, bf16* __restrict__ dst) {
  __shared__ float tile[32][33];
  const int r0 = blockIdx.y * 32, c0 = blockIdx.x * 32;
  const int t = threadIdx.x;
  const int rr = t >> 3, cc = (t & 7) << 2;
  float4 v = *(const float4*)(src + (size_t)(r0 + rr) * DMODEL + c0 + cc);
  tile[rr][cc + 0] = v.x; tile[rr][cc + 1] = v.y;
  tile[rr][cc + 2] = v.z; tile[rr][cc + 3] = v.w;
  __syncthreads();
  v4bf o;
  o[0] = (bf16)tile[cc + 0][rr];
  o[1] = (bf16)tile[cc + 1][rr];
  o[2] = (bf16)tile[cc + 2][rr];
  o[3] = (bf16)tile[cc + 3][rr];
  *(v4bf*)(dst + (size_t)(c0 + rr) * DMODEL + r0 + cc) = o;
}

// ---------------------------------------------------------------------------
// GEMM mainloop: C(128x128) += A(128xK) * Wt(NxK)^T, bf16 in, f32 acc.
// 8 waves: wave (wm 0..3, wn 0..1) owns 32x64 = 2x4 WMMA tiles.
// Async double-buffer: tile t+1 streams into LDS buf^1 while WMMAs run on
// buf; one s_wait_asynccnt + one barrier per K-step.
// ---------------------------------------------------------------------------
__device__ __forceinline__ void gemm_tile_core(
    const bf16* __restrict__ A, const bf16* __restrict__ Wt,
    int row0, int col0, int kdim,
    bf16 (*Alds)[128][40], bf16 (*Btlds)[128][40], v8f acc[2][4]) {
  const int tid  = threadIdx.x;
  const int lane = tid & 31, half = lane >> 4, l16 = lane & 15;
  const int wave = tid >> 5, wm = wave & 3, wn = wave >> 2;
  const int ra = tid >> 1, cba = (tid & 1) << 4;   // 128 rows x 32, 16 elem/thr

  const bf16* aptr = A  + (size_t)(row0 + ra) * DMODEL + cba;
  const bf16* bptr = Wt + (size_t)(col0 + ra) * DMODEL + cba;

  // prologue: stream tile 0 into buffer 0
  async_copy16(aptr,     &Alds[0][ra][cba]);
  async_copy16(aptr + 8, &Alds[0][ra][cba + 8]);
  async_copy16(bptr,     &Btlds[0][ra][cba]);
  async_copy16(bptr + 8, &Btlds[0][ra][cba + 8]);
  wait_async();
  __syncthreads();

  for (int k0 = 0; k0 < kdim; k0 += 32) {
    const int ib = (k0 >> 5) & 1;
    if (k0 + 32 < kdim) {   // stream next tile into the other buffer
      const bf16* ap = aptr + k0 + 32;
      const bf16* bp = bptr + k0 + 32;
      async_copy16(ap,     &Alds[ib ^ 1][ra][cba]);
      async_copy16(ap + 8, &Alds[ib ^ 1][ra][cba + 8]);
      async_copy16(bp,     &Btlds[ib ^ 1][ra][cba]);
      async_copy16(bp + 8, &Btlds[ib ^ 1][ra][cba + 8]);
    }

    v16bf afr[2];
#pragma unroll
    for (int i = 0; i < 2; ++i) {
      int r = wm * 32 + i * 16 + l16;
#pragma unroll
      for (int e = 0; e < 16; ++e) afr[i][e] = Alds[ib][r][frag_k(e, half)];
    }
    v16bf bfr[4];
#pragma unroll
    for (int n = 0; n < 4; ++n) {
      int c = wn * 64 + n * 16 + l16;
#pragma unroll
      for (int e = 0; e < 16; ++e) bfr[n][e] = Btlds[ib][c][frag_k(e, half)];
    }
#pragma unroll
    for (int i = 0; i < 2; ++i)
#pragma unroll
      for (int n = 0; n < 4; ++n)
        acc[i][n] = wmma_bf16(afr[i], bfr[n], acc[i][n]);

    wait_async();      // next tile landed in LDS
    __syncthreads();   // everyone done reading this tile + sees next tile
  }
}

// ---------------------------------------------------------------------------
// QKV projection (+RoPE on Q/K, +1/sqrt(Dh) on Q).
// Q,K -> (B,H,L,Dh) bf16.  V -> (B,H,Dh,L) bf16 (transposed, packed stores).
// grid: (D/128, B*L/128, 3)   z: 0=Q 1=K 2=V
// ---------------------------------------------------------------------------
__global__ __launch_bounds__(256) void qkv_gemm_kernel(
    const bf16* __restrict__ Xb,
    const bf16* __restrict__ Wqt, const bf16* __restrict__ Wkt,
    const bf16* __restrict__ Wvt,
    bf16* __restrict__ Qh, bf16* __restrict__ Kh, bf16* __restrict__ Vt) {
  __shared__ __align__(16) bf16 Alds[2][128][40];
  __shared__ __align__(16) bf16 Btlds[2][128][40];
  const int mode = blockIdx.z;
  const bf16* W = (mode == 0) ? Wqt : (mode == 1) ? Wkt : Wvt;

  const int row0 = blockIdx.y * 128;
  const int col0 = blockIdx.x * 128;

  v8f zero = {};
  v8f acc[2][4];
#pragma unroll
  for (int i = 0; i < 2; ++i)
#pragma unroll
    for (int n = 0; n < 4; ++n) acc[i][n] = zero;

  gemm_tile_core(Xb, W, row0, col0, DMODEL, Alds, Btlds, acc);

  const int tid = threadIdx.x, lane = tid & 31, half = lane >> 4, l16 = lane & 15;
  const int wave = tid >> 5, wm = wave & 3, wn = wave >> 2;

  if (mode == 2) {
    // V: 8 accumulator rows per lane are 8 consecutive positions l at fixed
    // head-dim -> one 16B packed store per tile per lane into (B,H,Dh,L).
#pragma unroll
    for (int i = 0; i < 2; ++i) {
#pragma unroll
      for (int n = 0; n < 4; ++n) {
        int col = col0 + wn * 64 + n * 16 + l16;
        int h = col >> 7, dh = col & 127;
        int lbase = row0 + wm * 32 + i * 16 + (half << 3);
        int b = lbase >> 11;
        int l = lbase & (L_SEQ - 1);
        v8bf t;
#pragma unroll
        for (int j = 0; j < 8; ++j) t[j] = (bf16)acc[i][n][j];
        *(v8bf*)(Vt + ((size_t)(b * NHEAD + h) * DHEAD + dh) * L_SEQ + l) = t;
      }
    }
    return;
  }

  bf16* dst = (mode == 0) ? Qh : Kh;
#pragma unroll
  for (int i = 0; i < 2; ++i) {
#pragma unroll
    for (int n = 0; n < 4; ++n) {
      int col = col0 + wn * 64 + n * 16 + l16;
      int h = col >> 7, dh = col & 127;
      float invf = fexp2(-((float)(dh & ~1) * (1.0f / 128.0f)) * LOG2_THETA);
      bool ev = (dh & 1) == 0;
#pragma unroll
      for (int j = 0; j < 8; ++j) {
        int grow = row0 + wm * 32 + i * 16 + j + (half << 3);
        int l = grow & (L_SEQ - 1);
        int b = grow >> 11;
        float v = acc[i][n][j];
        float ang = (float)l * invf;
        float cs = __cosf(ang), sn = __sinf(ang);
        float p = __shfl_xor(v, 1, 32);              // RoPE pair partner
        float r = ev ? (v * cs - p * sn) : (v * cs + p * sn);
        if (mode == 0) r *= QSCALE;
        dst[(((size_t)b * NHEAD + h) * L_SEQ + l) * DHEAD + dh] = (bf16)r;
      }
    }
  }
}

// ---------------------------------------------------------------------------
// Flash attention, causal.  grid: (L/128, B*H), block 256 (8 waves).
// Wave w owns 16 query rows; 32-key K/V tiles async-streamed into LDS
// (double-buffered), V dim-major.
// ---------------------------------------------------------------------------
__global__ __launch_bounds__(256) void attn_kernel(
    const bf16* __restrict__ Q, const bf16* __restrict__ K,
    const bf16* __restrict__ V, bf16* __restrict__ Attn) {
  __shared__ __align__(16) bf16 Klds[2][32][136];    // [buf][key][dim]
  __shared__ __align__(16) bf16 Vtlds[2][128][40];   // [buf][dim][key]
  __shared__ __align__(16) bf16 Plds[8][16][40];     // per-wave P transpose

  const int tid = threadIdx.x, wave = tid >> 5, lane = tid & 31;
  const int half = lane >> 4, l16 = lane & 15;
  const int bh = blockIdx.y;
  const bf16* q  = Q + (size_t)bh * L_SEQ * DHEAD;
  const bf16* kp = K + (size_t)bh * L_SEQ * DHEAD;
  const bf16* vp = V + (size_t)bh * DHEAD * L_SEQ;   // (Dh, L) per head
  const int qrow0 = blockIdx.x * 128 + wave * 16;

  // Q fragments: 4 x (16x32)
  v16bf qa[4];
#pragma unroll
  for (int kc = 0; kc < 4; ++kc)
#pragma unroll
    for (int e = 0; e < 16; ++e)
      qa[kc][e] = q[(size_t)(qrow0 + l16) * DHEAD + kc * 32 + frag_k(e, half)];

  v8f zero = {};
  v8f o[8];
#pragma unroll
  for (int d = 0; d < 8; ++d) o[d] = zero;
  float mst[8], lst[8];
#pragma unroll
  for (int j = 0; j < 8; ++j) { mst[j] = -3.0e38f; lst[j] = 0.0f; }

  const int rk = tid >> 3, cbk = (tid & 7) << 4;   // K tile 32x128
  const int rv = tid >> 1, cbv = (tid & 1) << 4;   // Vt tile 128x32
  const bf16* kbase = kp + (size_t)rk * DHEAD + cbk;
  const bf16* vbase = vp + (size_t)rv * L_SEQ + cbv;
  const int kvEnd = blockIdx.x * 128 + 128;

  // prologue: stream tile 0 into buffer 0
  async_copy16(kbase,     &Klds[0][rk][cbk]);
  async_copy16(kbase + 8, &Klds[0][rk][cbk + 8]);
  async_copy16(vbase,     &Vtlds[0][rv][cbv]);
  async_copy16(vbase + 8, &Vtlds[0][rv][cbv + 8]);
  wait_async();
  __syncthreads();

  for (int kv0 = 0; kv0 < kvEnd; kv0 += 32) {
    const int ib = (kv0 >> 5) & 1;
    if (kv0 + 32 < kvEnd) {   // stream next K/V tile into the other buffer
      const bf16* knxt = kbase + (size_t)(kv0 + 32) * DHEAD;
      const bf16* vnxt = vbase + (kv0 + 32);
      async_copy16(knxt,     &Klds[ib ^ 1][rk][cbk]);
      async_copy16(knxt + 8, &Klds[ib ^ 1][rk][cbk + 8]);
      async_copy16(vnxt,     &Vtlds[ib ^ 1][rv][cbv]);
      async_copy16(vnxt + 8, &Vtlds[ib ^ 1][rv][cbv + 8]);
    }

    if (kv0 <= qrow0 + 15) {   // wave-uniform causal skip (EXEC stays all-1)
      // S = Q * K^T, two 16-key groups
      v8f s[2];
      s[0] = zero; s[1] = zero;
#pragma unroll
      for (int nt = 0; nt < 2; ++nt)
#pragma unroll
        for (int kc = 0; kc < 4; ++kc) {
          v16bf bk;
#pragma unroll
          for (int e = 0; e < 16; ++e)
            bk[e] = Klds[ib][nt * 16 + l16][kc * 32 + frag_k(e, half)];
          s[nt] = wmma_bf16(qa[kc], bk, s[nt]);
        }

      // online softmax per row (row j+8*half spans 16 lanes of VGPR j)
#pragma unroll
      for (int j = 0; j < 8; ++j) {
        int grow = qrow0 + j + (half << 3);
        int key0 = kv0 + l16, key1 = kv0 + 16 + l16;
        float s0 = (key0 <= grow) ? s[0][j] : -3.0e38f;
        float s1 = (key1 <= grow) ? s[1][j] : -3.0e38f;
        float mx = fmaxf(s0, s1);
#pragma unroll
        for (int off = 1; off < 16; off <<= 1)
          mx = fmaxf(mx, __shfl_xor(mx, off, 32));
        float mnew = fmaxf(mst[j], mx);
        float p0 = fexp2((s0 - mnew) * LOG2E);
        float p1 = fexp2((s1 - mnew) * LOG2E);
        float rs = p0 + p1;
#pragma unroll
        for (int off = 1; off < 16; off <<= 1) rs += __shfl_xor(rs, off, 32);
        float alpha = fexp2((mst[j] - mnew) * LOG2E);
        lst[j] = lst[j] * alpha + rs;
        mst[j] = mnew;
#pragma unroll
        for (int d = 0; d < 8; ++d) o[d][j] *= alpha;
        Plds[wave][j + (half << 3)][l16]      = (bf16)p0;
        Plds[wave][j + (half << 3)][16 + l16] = (bf16)p1;
      }
      asm volatile("s_wait_dscnt 0" ::: "memory");  // P: C-layout -> A-layout

      // O += P(16x32) * V(32x128); V columns are contiguous Vt rows
      v16bf pa;
#pragma unroll
      for (int e = 0; e < 16; ++e) pa[e] = Plds[wave][l16][frag_k(e, half)];
#pragma unroll
      for (int d = 0; d < 8; ++d) {
        v16bf bv;
#pragma unroll
        for (int e = 0; e < 16; ++e)
          bv[e] = Vtlds[ib][d * 16 + l16][frag_k(e, half)];
        o[d] = wmma_bf16(pa, bv, o[d]);
      }
    }

    wait_async();      // next K/V tile landed
    __syncthreads();
  }

  // epilogue: O /= l, scatter to (B, L, H*Dh) bf16
  const int b = bh >> 4, h = bh & 15;
#pragma unroll
  for (int j = 0; j < 8; ++j) {
    int l = qrow0 + j + (half << 3);
    float inv = frcp(lst[j]);
#pragma unroll
    for (int d = 0; d < 8; ++d) {
      int col = h * DHEAD + d * 16 + l16;
      Attn[((size_t)b * L_SEQ + l) * DMODEL + col] = (bf16)(o[d][j] * inv);
    }
  }
}

// ---------------------------------------------------------------------------
// Output projection: out(f32) = Attn(bf16) * Wo, via Wo^T.  grid: (16, 32).
// ---------------------------------------------------------------------------
__global__ __launch_bounds__(256) void proj_gemm_kernel(
    const bf16* __restrict__ At, const bf16* __restrict__ Wot,
    float* __restrict__ Out) {
  __shared__ __align__(16) bf16 Alds[2][128][40];
  __shared__ __align__(16) bf16 Btlds[2][128][40];
  const int row0 = blockIdx.y * 128, col0 = blockIdx.x * 128;
  v8f zero = {};
  v8f acc[2][4];
#pragma unroll
  for (int i = 0; i < 2; ++i)
#pragma unroll
    for (int n = 0; n < 4; ++n) acc[i][n] = zero;

  gemm_tile_core(At, Wot, row0, col0, DMODEL, Alds, Btlds, acc);

  const int tid = threadIdx.x, lane = tid & 31, half = lane >> 4, l16 = lane & 15;
  const int wave = tid >> 5, wm = wave & 3, wn = wave >> 2;
#pragma unroll
  for (int i = 0; i < 2; ++i)
#pragma unroll
    for (int n = 0; n < 4; ++n) {
      int col = col0 + wn * 64 + n * 16 + l16;
#pragma unroll
      for (int j = 0; j < 8; ++j) {
        int grow = row0 + wm * 32 + i * 16 + j + (half << 3);
        Out[(size_t)grow * DMODEL + col] = acc[i][n][j];
      }
    }
}

// ---------------------------------------------------------------------------
extern "C" void kernel_launch(void* const* d_in, const int* in_sizes, int n_in,
                              void* d_out, int out_size, void* d_ws,
                              size_t ws_size, hipStream_t stream) {
  (void)in_sizes; (void)n_in; (void)out_size; (void)ws_size;
  const float* x  = (const float*)d_in[0];
  const float* Wq = (const float*)d_in[1];
  const float* Wk = (const float*)d_in[2];
  const float* Wv = (const float*)d_in[3];
  const float* Wo = (const float*)d_in[4];
  float* out = (float*)d_out;

  const size_t NX = (size_t)NBATCH * L_SEQ * DMODEL;  // 8,388,608
  const size_t NW = (size_t)DMODEL * DMODEL;          // 4,194,304
  char* w = (char*)d_ws;
  bf16* Xb  = (bf16*)w;  w += NX * 2;
  bf16* Wqt = (bf16*)w;  w += NW * 2;   // transposed bf16 weights
  bf16* Wkt = (bf16*)w;  w += NW * 2;
  bf16* Wvt = (bf16*)w;  w += NW * 2;
  bf16* Wot = (bf16*)w;  w += NW * 2;
  bf16* Qh  = (bf16*)w;  w += NX * 2;   // (B,H,L,Dh)
  bf16* Kh  = (bf16*)w;  w += NX * 2;   // (B,H,L,Dh)
  bf16* Vt  = (bf16*)w;  w += NX * 2;   // (B,H,Dh,L)
  bf16* At  = (bf16*)w;  w += NX * 2;   // (B,L,H*Dh)

  cvt_f32_bf16<<<(int)(NX / 4 / 256), 256, 0, stream>>>(x, Xb, (int)NX);
  dim3 tg(DMODEL / 32, DMODEL / 32);
  cvt_w_transpose<<<tg, 256, 0, stream>>>(Wq, Wqt);
  cvt_w_transpose<<<tg, 256, 0, stream>>>(Wk, Wkt);
  cvt_w_transpose<<<tg, 256, 0, stream>>>(Wv, Wvt);
  cvt_w_transpose<<<tg, 256, 0, stream>>>(Wo, Wot);

  qkv_gemm_kernel<<<dim3(DMODEL / 128, (NBATCH * L_SEQ) / 128, 3), 256, 0,
                    stream>>>(Xb, Wqt, Wkt, Wvt, Qh, Kh, Vt);

  attn_kernel<<<dim3(L_SEQ / 128, NBATCH * NHEAD), 256, 0, stream>>>(Qh, Kh, Vt,
                                                                     At);

  proj_gemm_kernel<<<dim3(DMODEL / 128, (NBATCH * L_SEQ) / 128), 256, 0,
                     stream>>>(At, Wot, out);
}
